// Graph_AE_19662360281451
// MI455X (gfx1250) — matrix operand order repaired and checked by reading, not compiled, not run on hardware
//
#include <hip/hip_runtime.h>

// ---------------------------------------------------------------------------
// Types / helpers
// ---------------------------------------------------------------------------
typedef __bf16 v16bf __attribute__((ext_vector_type(16)));
typedef float  v8f   __attribute__((ext_vector_type(8)));

#define NPOS 32768
#define CCH  512

static __device__ __forceinline__ unsigned short f2bf(float f) {
  union { float f; unsigned u; } x; x.f = f;
  unsigned r = x.u + 0x7FFFu + ((x.u >> 16) & 1u);   // round-to-nearest-even
  return (unsigned short)(r >> 16);
}
static __device__ __forceinline__ float bf2f(unsigned short u) {
  union { unsigned u; float f; } x; x.u = ((unsigned)u) << 16;
  return x.f;
}
static __device__ __forceinline__ v8f vzero() {
  v8f z;
#pragma unroll
  for (int i = 0; i < 8; ++i) z[i] = 0.f;
  return z;
}
static __device__ __forceinline__ v8f wmma_bf16(v16bf a, v16bf b, v8f c) {
  return __builtin_amdgcn_wmma_f32_16x16x32_bf16(false, a, false, b, (short)0, c,
                                                 false, false);
}

// A fragment (16x32 bf16): lane<16 holds K={0..15}, lane>=16 K={16..31};
// row M = lane&15.  One contiguous 16-elem run -> 2 x 16B vector loads.
static __device__ __forceinline__ v16bf load_a_bf(const unsigned short* __restrict__ W,
                                                  int stride, int mrow, int kbase,
                                                  int lane) {
  const unsigned short* p =
      W + (size_t)(mrow + (lane & 15)) * stride + kbase + ((lane & 16) ? 16 : 0);
  union { v16bf v; uint4 q[2]; } u;
  u.q[0] = *(const uint4*)(p);        // K = base .. base+7
  u.q[1] = *(const uint4*)(p + 8);    // K = base+8 .. base+15
  return u.v;
}

// B fragment (32x16 bf16) from an LDS tile stored TRANSPOSED as [n][k] with
// padded k-stride: lane's 16 elems are K = kbase + e (+16 if lane>=16) at fixed
// column n = nbase + (lane&15) -> contiguous in k -> 2 x ds_load_b128.
static __device__ __forceinline__ v16bf load_bT_lds(const unsigned short* sBT,
                                                    int kstride, int nbase, int kbase,
                                                    int lane) {
  const unsigned short* p =
      sBT + (size_t)(nbase + (lane & 15)) * kstride + kbase + ((lane & 16) ? 16 : 0);
  union { v16bf v; uint4 q[2]; } u;
  u.q[0] = *(const uint4*)(p);
  u.q[1] = *(const uint4*)(p + 8);
  return u.v;
}

// B fragment straight from global where k runs contiguously (used for V^T in k_ctx).
static __device__ __forceinline__ v16bf load_b_contig(const unsigned short* p) {
  union { v16bf v; uint4 q[2]; } u;
  u.q[0] = *(const uint4*)(p);
  u.q[1] = *(const uint4*)(p + 8);
  return u.v;
}

// ---------------------------------------------------------------------------
// 1) pack x,y [N,C] f32 -> inp_bf [2][C][N] bf16 (LDS tiled transpose)
// ---------------------------------------------------------------------------
__global__ void k_pack_inp(const float* __restrict__ x, const float* __restrict__ y,
                           unsigned short* __restrict__ inp) {
  __shared__ float tile[32][33];
  int b = blockIdx.z;
  const float* src = b ? y : x;
  int n0 = blockIdx.x * 32, c0 = blockIdx.y * 32;
  int tx = threadIdx.x, ty = threadIdx.y;  // 32 x 8
#pragma unroll
  for (int i = 0; i < 32; i += 8)
    tile[ty + i][tx] = src[(size_t)(n0 + ty + i) * CCH + c0 + tx];
  __syncthreads();
#pragma unroll
  for (int i = 0; i < 32; i += 8) {
    int c = c0 + ty + i, n = n0 + tx;
    inp[((size_t)b * CCH + c) * NPOS + n] = f2bf(tile[tx][ty + i]);
  }
}

// ---------------------------------------------------------------------------
// 2) weight conversion + BN folding
// ---------------------------------------------------------------------------
__global__ void k_convert_bf(const float* __restrict__ src,
                             unsigned short* __restrict__ dst, int n) {
  int i = blockIdx.x * 256 + threadIdx.x;
  if (i < n) dst[i] = f2bf(src[i]);
}

// A1 = diag(inv1) * adapt_w1 ; c1 = beta1 - mean1*inv1  (bn params: [gamma,beta,mean,var])
__global__ void k_bnfold1(const float* __restrict__ W1, const float* __restrict__ bn1,
                          float* __restrict__ A1, float* __restrict__ c1) {
  int i = blockIdx.x * 256 + threadIdx.x;  // 512*512
  int r = i >> 9, col = i & 511;
  float inv = bn1[r] * rsqrtf(bn1[3 * 512 + r] + 1e-5f);
  A1[i] = inv * W1[i];
  if (col == 0) c1[r] = bn1[512 + r] - bn1[2 * 512 + r] * inv;
}

// Wlow = diag(inv2) * adapt_w2 * A1  (bf16) ; blow = inv2*(W2@c1 - mean2) + beta2
__global__ void k_fold_low(const float* __restrict__ W2, const float* __restrict__ bn2,
                           const float* __restrict__ A1, const float* __restrict__ c1,
                           unsigned short* __restrict__ Wlow, float* __restrict__ blow) {
  int i = blockIdx.x * 256 + threadIdx.x;  // 256*512
  int o = i >> 9, j = i & 511;
  float inv2 = bn2[o] * rsqrtf(bn2[3 * 256 + o] + 1e-5f);
  float acc = 0.f;
  for (int c = 0; c < 512; ++c) acc += W2[o * 512 + c] * A1[c * 512 + j];
  Wlow[i] = f2bf(inv2 * acc);
  if (j == 0) {
    float s = 0.f;
    for (int c = 0; c < 512; ++c) s += W2[o * 512 + c] * c1[c];
    blow[o] = inv2 * (s - bn2[2 * 256 + o]) + bn2[256 + o];
  }
}

// ---------------------------------------------------------------------------
// 3) generic WMMA GEMM: out[b][O][N] = W[O][512] @ inp[b][512][N] + bias
//    block: 256 thr (8 waves), tile = 128 O x 64 N.
//    Activation tile staged TRANSPOSED in LDS: sBT[n][k], k-stride 520
//    (rows 16B-aligned, bank-spread) so B fragments are 2 x ds_load_b128.
// ---------------------------------------------------------------------------
#define KSTRIDE_G 520
__global__ __launch_bounds__(256) void k_gemm_bf(
    const unsigned short* __restrict__ W, const float* __restrict__ bias,
    const unsigned short* __restrict__ inp, unsigned short* __restrict__ out, int Otot) {
  __shared__ unsigned short sBT[64 * KSTRIDE_G];  // 65 KB
  int b = blockIdx.z;
  int nb = blockIdx.x * 64;
  int Ob = blockIdx.y * 128;
  const unsigned short* inpb = inp + (size_t)b * CCH * NPOS;
  unsigned short* outb = out + (size_t)b * Otot * NPOS;

  // stage transposed: read inp rows (contiguous n) as 16B vectors, scatter to [n][k]
  for (int i = threadIdx.x; i < 512 * 8; i += 256) {
    int c = i >> 3, g = (i & 7) * 8;
    uint4 d = *(const uint4*)(inpb + (size_t)c * NPOS + nb + g);
    const unsigned short* ds = (const unsigned short*)&d;
#pragma unroll
    for (int j = 0; j < 8; ++j) sBT[(size_t)(g + j) * KSTRIDE_G + c] = ds[j];
  }
  __syncthreads();

  int lane = threadIdx.x & 31, wave = threadIdx.x >> 5;
  int Mrow = Ob + wave * 16;
  __builtin_prefetch(W + (size_t)Mrow * 512, 0, 0);  // global_prefetch for weight strip

  v8f acc[4];
#pragma unroll
  for (int t = 0; t < 4; ++t) acc[t] = vzero();

  for (int ks = 0; ks < 16; ++ks) {
    v16bf a = load_a_bf(W, 512, Mrow, ks * 32, lane);
#pragma unroll
    for (int t = 0; t < 4; ++t) {
      v16bf bf = load_bT_lds(sBT, KSTRIDE_G, t * 16, ks * 32, lane);
      acc[t] = wmma_bf16(a, bf, acc[t]);
    }
  }

  int hi8 = (lane & 16) ? 8 : 0, nc = lane & 15;
#pragma unroll
  for (int t = 0; t < 4; ++t)
#pragma unroll
    for (int r = 0; r < 8; ++r) {
      int o = Mrow + r + hi8;
      int n = nb + t * 16 + nc;
      outb[(size_t)o * NPOS + n] = f2bf(acc[t][r] + bias[o]);
    }
}

// ---------------------------------------------------------------------------
// 4) tiny pool / delta-const path
// ---------------------------------------------------------------------------
__global__ void k_rowmean(const unsigned short* __restrict__ inp,
                          float* __restrict__ mean) {
  int row = blockIdx.x;  // 0..1023 = b*512+c
  __shared__ float red[256];
  const unsigned short* p = inp + (size_t)row * NPOS;
  float s = 0.f;
  for (int n = threadIdx.x; n < NPOS; n += 256) s += bf2f(p[n]);
  red[threadIdx.x] = s;
  __syncthreads();
  for (int st = 128; st; st >>= 1) {
    if ((int)threadIdx.x < st) red[threadIdx.x] += red[threadIdx.x + st];
    __syncthreads();
  }
  if (!threadIdx.x) mean[row] = red[0] * (1.f / NPOS);
}

__global__ void k_small(const float* __restrict__ mean, const float* __restrict__ pw1,
                        const float* __restrict__ pbn1, const float* __restrict__ pw2,
                        const float* __restrict__ pbn2, const float* __restrict__ dw1,
                        const float* __restrict__ dbn1, const float* __restrict__ dw2,
                        float* __restrict__ high, float* __restrict__ Mm,
                        float* __restrict__ Kc) {
  __shared__ float t1[2][512];
  int tid = threadIdx.x;
  for (int i = tid; i < 2 * 512; i += 256) {  // bn1(pool_w1 @ mean)
    int b = i >> 9, o = i & 511;
    float acc = 0.f;
    for (int c = 0; c < 512; ++c) acc += pw1[o * 512 + c] * mean[b * 512 + c];
    float inv = pbn1[o] * rsqrtf(pbn1[3 * 512 + o] + 1e-5f);
    t1[b][o] = (acc - pbn1[2 * 512 + o]) * inv + pbn1[512 + o];
  }
  __syncthreads();
  for (int i = tid; i < 2 * 256; i += 256) {  // high = bn2(pool_w2 @ t1)
    int b = i >> 8, o = i & 255;
    float acc = 0.f;
    for (int c = 0; c < 512; ++c) acc += pw2[o * 512 + c] * t1[b][c];
    float inv = pbn2[o] * rsqrtf(pbn2[3 * 256 + o] + 1e-5f);
    high[i] = (acc - pbn2[2 * 256 + o]) * inv + pbn2[256 + o];
  }
  for (int i = tid; i < 2 * 512; i += 256) {  // M = dw2 * diag(invd) * dw1  [2x512]
    int j = i >> 9, t = i & 511;
    float acc = 0.f;
    for (int f = 0; f < 256; ++f) {
      float inv = dbn1[f] * rsqrtf(dbn1[3 * 256 + f] + 1e-5f);
      acc += dw2[j * 256 + f] * inv * dw1[f * 512 + t];
    }
    Mm[i] = acc;
  }
  __syncthreads();
  if (tid < 4) {  // per-batch delta constants
    int b = tid >> 1, j = tid & 1;
    float acc = 0.f;
    for (int c = 0; c < 256; ++c) acc += Mm[j * 512 + 256 + c] * high[b * 256 + c];
    for (int f = 0; f < 256; ++f) {
      float inv = dbn1[f] * rsqrtf(dbn1[3 * 256 + f] + 1e-5f);
      acc += dw2[j * 256 + f] * (dbn1[256 + f] - dbn1[2 * 256 + f] * inv);
    }
    Kc[tid] = acc;
  }
}

// ---------------------------------------------------------------------------
// 5) softmax-K row max (512 rows of keys)
// ---------------------------------------------------------------------------
__global__ void k_rowmax(const unsigned short* __restrict__ kqv,
                         float* __restrict__ rowmax) {
  int r = blockIdx.x;  // b*256 + o
  int b = r >> 8, o = r & 255;
  const unsigned short* p = kqv + ((size_t)b * 768 + o) * NPOS;
  __shared__ float red[256];
  float m = -1e30f;
  for (int n = threadIdx.x; n < NPOS; n += 256) m = fmaxf(m, bf2f(p[n]));
  red[threadIdx.x] = m;
  __syncthreads();
  for (int st = 128; st; st >>= 1) {
    if ((int)threadIdx.x < st)
      red[threadIdx.x] = fmaxf(red[threadIdx.x], red[threadIdx.x + st]);
    __syncthreads();
  }
  if (!threadIdx.x) rowmax[r] = red[0];
}

// ---------------------------------------------------------------------------
// 6) ctx[b,h] = softmax(K) @ V^T via WMMA reduction over N ; store ctx^T bf16
// ---------------------------------------------------------------------------
__global__ __launch_bounds__(256) void k_ctx(const unsigned short* __restrict__ kqv,
                                             const float* __restrict__ rowmax,
                                             unsigned short* __restrict__ ctxT) {
  int b = blockIdx.x >> 3, h = blockIdx.x & 7;
  const unsigned short* keys = kqv + ((size_t)b * 768 + h * 32) * NPOS;
  const unsigned short* vals = kqv + ((size_t)b * 768 + 512 + h * 32) * NPOS;
  __shared__ float sCtx[32][32];
  __shared__ float sSum[32];
  for (int i = threadIdx.x; i < 32 * 32; i += 256) (&sCtx[0][0])[i] = 0.f;
  if (threadIdx.x < 32) sSum[threadIdx.x] = 0.f;
  __syncthreads();

  int lane = threadIdx.x & 31, wave = threadIdx.x >> 5;
  int khalf = (lane & 16) ? 8 : 0;
  int bhalf = (lane & 16) ? 16 : 0;
  float mrow[2];
  int rbase = b * 256 + h * 32;
  mrow[0] = rowmax[rbase + (lane & 15)];
  mrow[1] = rowmax[rbase + 16 + (lane & 15)];

  v8f acc[2][2];
#pragma unroll
  for (int tm = 0; tm < 2; ++tm)
#pragma unroll
    for (int tn = 0; tn < 2; ++tn) acc[tm][tn] = vzero();
  float psum[2] = {0.f, 0.f};

  for (int ncb = wave * 32; ncb < NPOS; ncb += 256) {
    v16bf afrag[2], bfrag[2];
#pragma unroll
    for (int tm = 0; tm < 2; ++tm) {  // A = exp(keys - max), built on the fly
      union { v16bf v; unsigned short s[16]; } u;
      const unsigned short* kp = keys + (size_t)(tm * 16 + (lane & 15)) * NPOS + ncb;
      float m = mrow[tm];
#pragma unroll
      for (int e = 0; e < 16; ++e) {
        float v = __expf(bf2f(kp[e + bhalf]) - m);
        psum[tm] += v;
        u.s[e] = f2bf(v);
      }
      afrag[tm] = u.v;
    }
#pragma unroll
    for (int tn = 0; tn < 2; ++tn)  // B = V^T (k=n contiguous) -> 2 x b128
      bfrag[tn] = load_b_contig(vals + (size_t)(tn * 16 + (lane & 15)) * NPOS + ncb +
                                bhalf);
#pragma unroll
    for (int tm = 0; tm < 2; ++tm)
#pragma unroll
      for (int tn = 0; tn < 2; ++tn)
        acc[tm][tn] = wmma_bf16(afrag[tm], bfrag[tn], acc[tm][tn]);
  }

  int hi8 = (lane & 16) ? 8 : 0, nidx = lane & 15;
#pragma unroll
  for (int tm = 0; tm < 2; ++tm) {
    atomicAdd(&sSum[tm * 16 + (lane & 15)], psum[tm]);
#pragma unroll
    for (int tn = 0; tn < 2; ++tn)
#pragma unroll
      for (int r = 0; r < 8; ++r)
        atomicAdd(&sCtx[tm * 16 + r + hi8][tn * 16 + nidx], acc[tm][tn][r]);
  }
  __syncthreads();
  unsigned short* dst = ctxT + (size_t)(b * 8 + h) * 1024;
  for (int i = threadIdx.x; i < 32 * 32; i += 256) {
    int v = i >> 5, k = i & 31;
    dst[i] = f2bf(sCtx[k][v] / sSum[k]);  // stored transposed [v][k]
  }
}

// ---------------------------------------------------------------------------
// 7) att[b, h*32+v, n] = ctx^T @ softmax_channels(Q)  (fused Q-softmax)
// ---------------------------------------------------------------------------
__global__ __launch_bounds__(256) void k_att(const unsigned short* __restrict__ kqv,
                                             const unsigned short* __restrict__ ctxT,
                                             unsigned short* __restrict__ att) {
  int b = blockIdx.z, h = blockIdx.y;
  int nb = blockIdx.x * 128;
  int lane = threadIdx.x & 31, wave = threadIdx.x >> 5;
  int n = nb + wave * 16 + (lane & 15);
  const unsigned short* q = kqv + ((size_t)b * 768 + 256 + h * 32) * NPOS + n;

  float qv[32];
  float m = -1e30f;
#pragma unroll
  for (int k = 0; k < 32; ++k) {
    qv[k] = bf2f(q[(size_t)k * NPOS]);
    m = fmaxf(m, qv[k]);
  }
  float s = 0.f;
#pragma unroll
  for (int k = 0; k < 32; ++k) {
    qv[k] = __expf(qv[k] - m);
    s += qv[k];
  }
  float inv = 1.f / s;

  union { v16bf v; unsigned short sb[16]; } ub;
  int bhalf = (lane & 16) ? 16 : 0;
#pragma unroll
  for (int e = 0; e < 16; ++e) ub.sb[e] = f2bf(qv[e + bhalf] * inv);

  const unsigned short* cT = ctxT + (size_t)(b * 8 + h) * 1024;
  int hi8 = (lane & 16) ? 8 : 0;
#pragma unroll
  for (int tm = 0; tm < 2; ++tm) {
    v16bf a = load_a_bf(cT, 32, tm * 16, 0, lane);
    v8f d = wmma_bf16(a, ub.v, vzero());
#pragma unroll
    for (int r = 0; r < 8; ++r) {
      int vc = tm * 16 + r + hi8;
      att[((size_t)b * 256 + h * 32 + vc) * NPOS + n] = f2bf(d[r]);
    }
  }
}

// ---------------------------------------------------------------------------
// 8) grid-sample weight: delta = M @ low + Kc, wgt = bilinear zero-pad weight
// ---------------------------------------------------------------------------
__global__ void k_wgt(const unsigned short* __restrict__ low,
                      const float* __restrict__ Mm, const float* __restrict__ Kc,
                      float* __restrict__ wgt) {
  int t = blockIdx.x * 256 + threadIdx.x;  // 2*N
  int b = t >> 15, n = t & (NPOS - 1);
  const unsigned short* lp = low + (size_t)b * 256 * NPOS + n;
  float d0 = Kc[b * 2 + 0], d1 = Kc[b * 2 + 1];
  for (int c = 0; c < 256; ++c) {
    float lv = bf2f(lp[(size_t)c * NPOS]);
    d0 += Mm[c] * lv;
    d1 += Mm[512 + c] * lv;
  }
  float gx = -1.f + 2.f * (float)n / (float)(NPOS - 1) + d0;
  float gy = -1.f + d1;
  float ix = gx * 0.5f, iy = gy * 0.5f;
  float w = fmaxf(0.f, fminf(1.f, 1.f - fabsf(ix))) *
            fmaxf(0.f, fminf(1.f, 1.f - fabsf(iy)));
  wgt[t] = w;
}

// ---------------------------------------------------------------------------
// 9) fused output: out[n,O] = 0.25*Sum_b(Wr@att_b + br + inp_b) + 0.25*Sum_b cam_b
//    att tile staged transposed [n][k], k-stride 264 -> 2 x ds_load_b128 frags
// ---------------------------------------------------------------------------
#define KSTRIDE_O 264
__global__ __launch_bounds__(256) void k_out(
    const unsigned short* __restrict__ Wr, const float* __restrict__ br,
    const unsigned short* __restrict__ att, const unsigned short* __restrict__ inp,
    const unsigned short* __restrict__ low, const float* __restrict__ high,
    const float* __restrict__ wgt, float* __restrict__ out) {
  __shared__ unsigned short sBT[64 * KSTRIDE_O];  // 33 KB
  int nb = blockIdx.x * 64, Ob = blockIdx.y * 128;
  int lane = threadIdx.x & 31, wave = threadIdx.x >> 5;
  int Mrow = Ob + wave * 16;

  v8f acc[4];
#pragma unroll
  for (int t = 0; t < 4; ++t) acc[t] = vzero();

  for (int b = 0; b < 2; ++b) {
    __syncthreads();
    for (int i = threadIdx.x; i < 256 * 8; i += 256) {
      int c = i >> 3, g = (i & 7) * 8;
      uint4 d = *(const uint4*)(att + ((size_t)b * 256 + c) * NPOS + nb + g);
      const unsigned short* ds = (const unsigned short*)&d;
#pragma unroll
      for (int j = 0; j < 8; ++j) sBT[(size_t)(g + j) * KSTRIDE_O + c] = ds[j];
    }
    __syncthreads();
    for (int ks = 0; ks < 8; ++ks) {
      v16bf a = load_a_bf(Wr, 256, Mrow, ks * 32, lane);
#pragma unroll
      for (int t = 0; t < 4; ++t) {
        v16bf bf = load_bT_lds(sBT, KSTRIDE_O, t * 16, ks * 32, lane);
        acc[t] = wmma_bf16(a, bf, acc[t]);  // accumulates across both batches
      }
    }
  }

  int hi8 = (lane & 16) ? 8 : 0, nc = lane & 15;
#pragma unroll
  for (int t = 0; t < 4; ++t)
#pragma unroll
    for (int r = 0; r < 8; ++r) {
      int O = Mrow + r + hi8;
      int n = nb + t * 16 + nc;
      float ea = acc[t][r] + 2.f * br[O] +
                 bf2f(inp[(size_t)O * NPOS + n]) +
                 bf2f(inp[((size_t)512 + O) * NPOS + n]);
      float cam;
      if (O < 256) {
        cam = high[O] * wgt[n] + high[256 + O] * wgt[NPOS + n];
      } else {
        int c = O - 256;
        cam = bf2f(low[(size_t)c * NPOS + n]) +
              bf2f(low[((size_t)256 + c) * NPOS + n]);
      }
      out[(size_t)n * CCH + O] = 0.25f * ea + 0.25f * cam;
    }
}

// ---------------------------------------------------------------------------
// launcher
// ---------------------------------------------------------------------------
extern "C" void kernel_launch(void* const* d_in, const int* in_sizes, int n_in,
                              void* d_out, int out_size, void* d_ws, size_t ws_size,
                              hipStream_t stream) {
  (void)in_sizes; (void)n_in; (void)out_size; (void)ws_size;
  const float* x        = (const float*)d_in[0];
  const float* y        = (const float*)d_in[1];
  const float* ea_wk    = (const float*)d_in[2];
  const float* ea_bk    = (const float*)d_in[3];
  const float* ea_wq    = (const float*)d_in[4];
  const float* ea_bq    = (const float*)d_in[5];
  const float* ea_wv    = (const float*)d_in[6];
  const float* ea_bv    = (const float*)d_in[7];
  const float* ea_wr    = (const float*)d_in[8];
  const float* ea_br    = (const float*)d_in[9];
  const float* pool_w1  = (const float*)d_in[10];
  const float* pool_bn1 = (const float*)d_in[11];
  const float* pool_w2  = (const float*)d_in[12];
  const float* pool_bn2 = (const float*)d_in[13];
  const float* adapt_w1 = (const float*)d_in[14];
  const float* adapt_bn1= (const float*)d_in[15];
  const float* adapt_w2 = (const float*)d_in[16];
  const float* adapt_bn2= (const float*)d_in[17];
  const float* delta_w1 = (const float*)d_in[18];
  const float* delta_bn1= (const float*)d_in[19];
  const float* delta_w2 = (const float*)d_in[20];

  char* ws = (char*)d_ws;
  size_t off = 0;
  auto alloc = [&](size_t bytes) -> void* {
    void* p = ws + off;
    off += (bytes + 255) & ~(size_t)255;
    return p;
  };
  unsigned short* inp_bf = (unsigned short*)alloc(2ull * CCH * NPOS * 2);
  unsigned short* kqv    = (unsigned short*)alloc(2ull * 768 * NPOS * 2);
  unsigned short* low_bf = (unsigned short*)alloc(2ull * 256 * NPOS * 2);
  unsigned short* att_bf = (unsigned short*)alloc(2ull * 256 * NPOS * 2);
  unsigned short* Wkqv   = (unsigned short*)alloc(768 * 512 * 2);
  float*          bkqv   = (float*)alloc(768 * 4);
  unsigned short* Wrb    = (unsigned short*)alloc(512 * 256 * 2);
  unsigned short* Wlow   = (unsigned short*)alloc(256 * 512 * 2);
  float*          blow   = (float*)alloc(256 * 4);
  float*          A1     = (float*)alloc(512ull * 512 * 4);
  float*          c1     = (float*)alloc(512 * 4);
  float*          meanv  = (float*)alloc(1024 * 4);
  float*          high   = (float*)alloc(512 * 4);
  float*          Mm     = (float*)alloc(1024 * 4);
  float*          Kc     = (float*)alloc(4 * 4);
  float*          rowmax = (float*)alloc(512 * 4);
  unsigned short* ctxT   = (unsigned short*)alloc(2 * 8 * 32 * 32 * 2);
  float*          wgt    = (float*)alloc(2ull * NPOS * 4);

  // 1) pack input
  k_pack_inp<<<dim3(NPOS / 32, CCH / 32, 2), dim3(32, 8), 0, stream>>>(x, y, inp_bf);

  // 2) weights: convert + fold BN
  k_convert_bf<<<(256 * 512) / 256, 256, 0, stream>>>(ea_wk, Wkqv, 256 * 512);
  k_convert_bf<<<(256 * 512) / 256, 256, 0, stream>>>(ea_wq, Wkqv + 256 * 512, 256 * 512);
  k_convert_bf<<<(256 * 512) / 256, 256, 0, stream>>>(ea_wv, Wkqv + 512 * 512, 256 * 512);
  k_convert_bf<<<(512 * 256) / 256, 256, 0, stream>>>(ea_wr, Wrb, 512 * 256);
  hipMemcpyAsync(bkqv,       ea_bk, 256 * 4, hipMemcpyDeviceToDevice, stream);
  hipMemcpyAsync(bkqv + 256, ea_bq, 256 * 4, hipMemcpyDeviceToDevice, stream);
  hipMemcpyAsync(bkqv + 512, ea_bv, 256 * 4, hipMemcpyDeviceToDevice, stream);
  k_bnfold1<<<(512 * 512) / 256, 256, 0, stream>>>(adapt_w1, adapt_bn1, A1, c1);
  k_fold_low<<<(256 * 512) / 256, 256, 0, stream>>>(adapt_w2, adapt_bn2, A1, c1, Wlow, blow);

  // 3) big WMMA GEMMs
  k_gemm_bf<<<dim3(NPOS / 64, 6, 2), 256, 0, stream>>>(Wkqv, bkqv, inp_bf, kqv, 768);
  k_gemm_bf<<<dim3(NPOS / 64, 2, 2), 256, 0, stream>>>(Wlow, blow, inp_bf, low_bf, 256);

  // 4) pool + delta constants
  k_rowmean<<<1024, 256, 0, stream>>>(inp_bf, meanv);
  k_small<<<1, 256, 0, stream>>>(meanv, pool_w1, pool_bn1, pool_w2, pool_bn2,
                                 delta_w1, delta_bn1, delta_w2, high, Mm, Kc);

  // 5-7) attention core
  k_rowmax<<<512, 256, 0, stream>>>(kqv, rowmax);
  k_ctx<<<16, 256, 0, stream>>>(kqv, rowmax, ctxT);
  k_att<<<dim3(NPOS / 128, 8, 2), 256, 0, stream>>>(kqv, ctxT, att_bf);

  // 8) CAM grid-sample weights
  k_wgt<<<(2 * NPOS) / 256, 256, 0, stream>>>(low_bf, Mm, Kc, wgt);

  // 9) fused reprojection + residual + CAM epilogue -> out [N, C] fp32
  k_out<<<dim3(NPOS / 64, 4), 256, 0, stream>>>(Wrb, ea_br, att_bf, inp_bf, low_bf,
                                                high, wgt, (float*)d_out);
}
